// ScaleDotProductAttention_30562987278888
// MI455X (gfx1250) — matrix operand order repaired and checked by reading, not compiled, not run on hardware
//
#include <hip/hip_runtime.h>
#include <hip/hip_bf16.h>

// Fused scaled-dot-product attention for (b=2, h=16, n=2048, d=64), fp32 I/O.
// d_out = [ out : 2*16*2048*64 f32 | attn : 2*16*2048*2048 f32 ].
// bf16 WMMA (v_wmma_f32_16x16x32_bf16) for both GEMMs, f32 softmax.
// attn/out stores are non-temporal so the L2-resident K/V working set
// (32 MB << 192 MB L2) is not flushed by the 537 MB probability stream.

typedef __attribute__((ext_vector_type(16))) __bf16 v16bf;
typedef __attribute__((ext_vector_type(8)))  float  v8f;

// may_alias types: all type-punned accesses go through these so the staging
// stores (bf16) and fragment loads (u32) are correctly ordered by the compiler.
typedef unsigned int __attribute__((may_alias)) u32a;
typedef __bf16       __attribute__((may_alias)) bf16a;

#define N_TOK 2048
#define D_K   64
#define QBLK  16      // query rows per workgroup
#define NWAVE 8       // 256 threads = 8 wave32
#define SROW  2052    // padded f32 row stride for score buffer (2048+4)
#define QSTR  66      // padded bf16 row stride for 64-wide tiles (64+2)
#define VSTR  18      // padded bf16 row stride for 16-wide V slices (16+2)

union Frag {
    v16bf v;
    unsigned int u[8];
};

__global__ __launch_bounds__(256, 1)
void sdpa_fused_kernel(const float* __restrict__ Q,
                       const float* __restrict__ K,
                       const float* __restrict__ V,
                       float* __restrict__ Out,     // [32, 2048, 64]
                       float* __restrict__ Attn)    // [32, 2048, 2048]
{
    __shared__ __bf16 Qs[QBLK * QSTR];            //   2.1 KB : Q tile (scaled, bf16)
    __shared__ __bf16 Kt[NWAVE][16 * QSTR];       //  16.9 KB : per-wave K tiles
    __shared__ __bf16 Vt[NWAVE][32 * VSTR];       //   9.2 KB : per-wave V slices
    __shared__ float  Sb[QBLK * SROW];            // 128.3 KB : scores / probs
    __shared__ float  Ored[4][32][8];             //   4.0 KB : partial-O reduction

    const int tid  = threadIdx.x;
    const int lane = tid & 31;
    const int wave = tid >> 5;
    const int g    = lane >> 4;   // lane half-group (0/1) per WMMA layouts
    const int nl   = lane & 15;

    const int qblk = blockIdx.x & 127;   // 128 q-blocks of 16 rows
    const int bh   = blockIdx.x >> 7;    // 32 (batch, head) pairs
    const float scale = 0.125f;          // 1/sqrt(64)

    const float* Qg = Q + ((size_t)bh * N_TOK + (size_t)qblk * QBLK) * D_K;
    const float* Kg = K + (size_t)bh * N_TOK * D_K;
    const float* Vg = V + (size_t)bh * N_TOK * D_K;

    // ---------------- Phase 0: stage Q tile (scale folded in) ----------------
    {
        const int idx = tid * 4;                  // 16*64 = 1024 f32 / 256 thr
        const int r = idx >> 6, c = idx & 63;
        const float4 q4 = *(const float4*)(Qg + r * D_K + c);
        __bf16* dst = &Qs[r * QSTR + c];
        dst[0] = (__bf16)(q4.x * scale);
        dst[1] = (__bf16)(q4.y * scale);
        dst[2] = (__bf16)(q4.z * scale);
        dst[3] = (__bf16)(q4.w * scale);
    }
    __syncthreads();

    // A fragments of Q (ISA 16-bit A 16x32 layout), loop-invariant.
    Frag a0, a1;
#pragma unroll
    for (int i = 0; i < 8; ++i) {
        const int kk = (i < 4) ? (g * 8 + 2 * i) : (16 + g * 8 + 2 * (i - 4));
        a0.u[i] = *(const u32a*)&Qs[nl * QSTR + kk];
        a1.u[i] = *(const u32a*)&Qs[nl * QSTR + 32 + kk];
    }

    // ---------------- Phase 1: S = (Q*scale) K^T via WMMA ----------------
    for (int j = wave; j < N_TOK / 16; j += NWAVE) {
        __bf16* kt = Kt[wave];
#pragma unroll
        for (int it = 0; it < 8; ++it) {          // 16x64 f32 tile -> bf16 LDS
            const int idx = it * 128 + lane * 4;
            const int r = idx >> 6, c = idx & 63;
            const float4 k4 =
                *(const float4*)(Kg + (size_t)(j * 16 + r) * D_K + c);
            __bf16* dst = &kt[r * QSTR + c];
            dst[0] = (__bf16)k4.x; dst[1] = (__bf16)k4.y;
            dst[2] = (__bf16)k4.z; dst[3] = (__bf16)k4.w;
        }
        // B fragments (ISA 16-bit B 32x16 layout): B[d][n] = Ktile[n][d]
        Frag b0, b1;
#pragma unroll
        for (int i = 0; i < 8; ++i) {
            const int d0 = (i < 4) ? (g * 8 + 2 * i) : (16 + g * 8 + 2 * (i - 4));
            b0.u[i] = *(const u32a*)&kt[nl * QSTR + d0];
            b1.u[i] = *(const u32a*)&kt[nl * QSTR + 32 + d0];
        }
        v8f c = {};
        c = __builtin_amdgcn_wmma_f32_16x16x32_bf16(false, a0.v, false, b0.v,
                                                    (short)0, c, false, false);
        c = __builtin_amdgcn_wmma_f32_16x16x32_bf16(false, a1.v, false, b1.v,
                                                    (short)0, c, false, false);
        // C layout: VGPR i, lanes 0-15 -> M=i, lanes 16-31 -> M=8+i, N=lane%16
#pragma unroll
        for (int i = 0; i < 8; ++i)
            Sb[(i + 8 * g) * SROW + j * 16 + nl] = c[i];
    }
    __syncthreads();

    // ---------------- Phase 2: softmax + write attn + pack P (bf16) ----------
#pragma unroll
    for (int rr = 0; rr < 2; ++rr) {
        const int r = 2 * wave + rr;
        float* srow = &Sb[r * SROW];

        float m = -INFINITY;
        for (int t = 0; t < N_TOK / 32; ++t)
            m = fmaxf(m, srow[lane + 32 * t]);
#pragma unroll
        for (int off = 16; off > 0; off >>= 1)
            m = fmaxf(m, __shfl_xor(m, off, 32));

        float s = 0.f;
        for (int t = 0; t < N_TOK / 32; ++t) {
            const int c = lane + 32 * t;
            const float e = __expf(srow[c] - m);
            srow[c] = e;
            s += e;
        }
#pragma unroll
        for (int off = 16; off > 0; off >>= 1)
            s += __shfl_xor(s, off, 32);
        const float inv = 1.0f / s;

        float* arow = Attn + ((size_t)bh * N_TOK + (size_t)qblk * QBLK + r) * N_TOK;
        bf16a* prow = (bf16a*)srow;   // in-place pack: write@2c < read@4c, safe
        for (int t = 0; t < N_TOK / 32; ++t) {
            const int c = lane + 32 * t;
            const float p = srow[c] * inv;
            __builtin_nontemporal_store(p, &arow[c]);  // streaming 537MB, keep L2
            prow[c] = (__bf16)p;
        }
    }
    __syncthreads();

    // ---------------- Phase 3: O = P V via WMMA ----------------
    const int nt   = wave & 3;   // which 16-wide slice of d
    const int half = wave >> 2;  // which half of the k range
    v8f oc = {};
    __bf16* vt = Vt[wave];

    for (int ks = half * 32; ks < half * 32 + 32; ++ks) {
        const int kbase = ks * 32;                 // token-row base
#pragma unroll
        for (int it = 0; it < 4; ++it) {           // stage 32x16 V slice as bf16
            const int idx = it * 128 + lane * 4;   // 512 f32
            const int r = idx >> 4, c = idx & 15;
            const float4 v4 =
                *(const float4*)(Vg + (size_t)(kbase + r) * D_K + nt * 16 + c);
            __bf16* dst = &vt[r * VSTR + c];
            dst[0] = (__bf16)v4.x; dst[1] = (__bf16)v4.y;
            dst[2] = (__bf16)v4.z; dst[3] = (__bf16)v4.w;
        }
        // A fragment from packed P rows (bf16 pairs are 4B-aligned: kk even)
        Frag pa;
#pragma unroll
        for (int i = 0; i < 8; ++i) {
            const int kk = (i < 4) ? (g * 8 + 2 * i) : (16 + g * 8 + 2 * (i - 4));
            pa.u[i] = *(const u32a*)((const bf16a*)&Sb[nl * SROW] + kbase + kk);
        }
        // B fragment: B[kk][n] = Vslice[kk][n]; rows kk,kk+1 -> two b16 loads
        Frag vb;
#pragma unroll
        for (int i = 0; i < 8; ++i) {
            const int kk = (i < 4) ? (g * 8 + 2 * i) : (16 + g * 8 + 2 * (i - 4));
            vb.v[2 * i]     = vt[kk * VSTR + nl];
            vb.v[2 * i + 1] = vt[(kk + 1) * VSTR + nl];
        }
        oc = __builtin_amdgcn_wmma_f32_16x16x32_bf16(false, pa.v, false, vb.v,
                                                     (short)0, oc, false, false);
    }

    // Reduce the two k-halves through LDS, then write O.
    if (half == 1) {
#pragma unroll
        for (int i = 0; i < 8; ++i) Ored[nt][lane][i] = oc[i];
    }
    __syncthreads();
    if (half == 0) {
#pragma unroll
        for (int i = 0; i < 8; ++i) oc[i] += Ored[nt][lane][i];
        float* og = Out + ((size_t)bh * N_TOK + (size_t)qblk * QBLK) * D_K;
#pragma unroll
        for (int i = 0; i < 8; ++i)
            __builtin_nontemporal_store(oc[i],
                &og[(size_t)(i + 8 * g) * D_K + nt * 16 + nl]);
    }
}

extern "C" void kernel_launch(void* const* d_in, const int* in_sizes, int n_in,
                              void* d_out, int out_size, void* d_ws, size_t ws_size,
                              hipStream_t stream) {
    (void)in_sizes; (void)n_in; (void)out_size; (void)d_ws; (void)ws_size;
    const float* Q = (const float*)d_in[0];
    const float* K = (const float*)d_in[1];
    const float* V = (const float*)d_in[2];
    float* Out  = (float*)d_out;
    float* Attn = Out + (size_t)2 * 16 * 2048 * 64;   // attn follows out

    const int grid = 32 * (N_TOK / QBLK);             // 4096 workgroups
    sdpa_fused_kernel<<<grid, 256, 0, stream>>>(Q, K, V, Out, Attn);
}
